// MHSAttentionBlock_51427938402678
// MI455X (gfx1250) — matrix operand order repaired and checked by reading, not compiled, not run on hardware
//
#include <hip/hip_runtime.h>
#include <hip/hip_bf16.h>

typedef __attribute__((ext_vector_type(16))) _Float16 v16h;
typedef __attribute__((ext_vector_type(8)))  _Float16 v8h;
typedef __attribute__((ext_vector_type(8)))  float    v8f;

union V16 { v16h v; v8h h[2]; };

// A-fragment (16x32 f16, row-major source, per-lane row = lane%16):
// lane<16: K = k0+[0..7] and k0+16+[0..7]; lane>=16: +8 on both chunks.
__device__ __forceinline__ v16h load_a(const _Float16* row, int k0, int lg) {
  V16 u;
  u.h[0] = *(const v8h*)(row + k0 + lg * 8);
  u.h[1] = *(const v8h*)(row + k0 + 16 + lg * 8);
  return u.v;
}
// B-fragment (32x16 f16, n-major source, per-lane col = lane%16):
// lane<16: K = k0+[0..7] and k0+16+[0..7]; lane>=16: +8 on both chunks.
__device__ __forceinline__ v16h load_b(const _Float16* col, int k0, int lg) {
  V16 u;
  u.h[0] = *(const v8h*)(col + k0 + lg * 8);
  u.h[1] = *(const v8h*)(col + k0 + 16 + lg * 8);
  return u.v;
}

#define WMMA_F16(a, b, c) \
  __builtin_amdgcn_wmma_f32_16x16x32_f16(false, (a), false, (b), (short)0, (c), false, false)

// ---------------- Kernel 0: weight f32 -> f16 conversion ----------------
__global__ void cvt_kernel(const float* __restrict__ wq, const float* __restrict__ wo,
                           _Float16* __restrict__ wq_h, _Float16* __restrict__ wo_h) {
  int i = blockIdx.x * 256 + threadIdx.x;
  if (i < 1536 * 512) wq_h[i] = (_Float16)wq[i];
  if (i < 512 * 512)  wo_h[i] = (_Float16)wo[i];
}

// ---------------- Kernel 1: GroupNorm -> xn_t[b][hw][c] (f16) ----------------
__global__ __launch_bounds__(256) void gn_kernel(const float* __restrict__ x,
                                                 const float* __restrict__ gamma,
                                                 const float* __restrict__ beta,
                                                 _Float16* __restrict__ xn_t) {
  int bg = blockIdx.x;
  int b = bg >> 5, g = bg & 31;
  const float* base = x + (size_t)(b * 512 + g * 16) * 1024;
  int t = threadIdx.x;
  float s = 0.f, ss = 0.f;
  for (int idx = t; idx < 16 * 1024; idx += 256) {
    float v = base[idx];
    s += v; ss += v * v;
  }
  __shared__ float rs[256], rss[256];
  __shared__ float s_mean, s_rstd;
  rs[t] = s; rss[t] = ss;
  __syncthreads();
  for (int off = 128; off > 0; off >>= 1) {
    if (t < off) { rs[t] += rs[t + off]; rss[t] += rss[t + off]; }
    __syncthreads();
  }
  if (t == 0) {
    float mean = rs[0] * (1.0f / 16384.0f);
    float var  = rss[0] * (1.0f / 16384.0f) - mean * mean;
    s_mean = mean;
    s_rstd = rsqrtf(var + 1e-5f);
  }
  __syncthreads();
  float mean = s_mean, rstd = s_rstd;
  for (int idx = t; idx < 16 * 1024; idx += 256) {
    int cl = idx >> 10, i = idx & 1023;
    int c = g * 16 + cl;
    float v = (base[idx] - mean) * rstd * gamma[c] + beta[c];
    xn_t[(size_t)(b * 1024 + i) * 512 + c] = (_Float16)v;
  }
}

// ---------------- Kernel 2: QKV GEMM, 32x32 tile per wave (2x2 WMMA blocking) ----------------
// q_t,k_t: [b*8+head][hw][64]  v: [b*8+head][64][hw]
__global__ __launch_bounds__(256) void qkv_kernel(const _Float16* __restrict__ wq_h,
                                                  const float* __restrict__ b_qkv,
                                                  const _Float16* __restrict__ xn_t,
                                                  _Float16* __restrict__ q_t,
                                                  _Float16* __restrict__ k_t,
                                                  _Float16* __restrict__ v_m) {
  int lane = threadIdx.x & 31, wave = threadIdx.x >> 5;
  int g = blockIdx.x * 8 + wave;             // 12288 wave-tiles total
  int b = g / 1536, rem = g % 1536;          // 48 x 32 tiles of 32x32 per batch
  int mt = rem >> 5, nt = rem & 31;
  int lg = lane >> 4, ln = lane & 15;
  const _Float16* arow0 = wq_h + (size_t)(mt * 32 + ln) * 512;
  const _Float16* arow1 = arow0 + (size_t)16 * 512;
  const _Float16* bcol0 = xn_t + (size_t)(b * 1024 + nt * 32 + ln) * 512;
  const _Float16* bcol1 = bcol0 + (size_t)16 * 512;
  v8f acc00 = {}, acc01 = {}, acc10 = {}, acc11 = {};
  for (int k0 = 0; k0 < 512; k0 += 32) {
    v16h a0 = load_a(arow0, k0, lg);
    v16h a1 = load_a(arow1, k0, lg);
    v16h b0 = load_b(bcol0, k0, lg);
    v16h b1 = load_b(bcol1, k0, lg);
    acc00 = WMMA_F16(a0, b0, acc00);
    acc01 = WMMA_F16(a0, b1, acc01);
    acc10 = WMMA_F16(a1, b0, acc10);
    acc11 = WMMA_F16(a1, b1, acc11);
  }
  for (int half = 0; half < 4; ++half) {
    const v8f& acc = half == 0 ? acc00 : half == 1 ? acc01 : half == 2 ? acc10 : acc11;
    int mo = (half >> 1) * 16, no = (half & 1) * 16;
    int i = nt * 32 + no + ln;
    for (int r = 0; r < 8; ++r) {
      int m = r + lg * 8;
      int o = mt * 32 + mo + m;
      _Float16 hv = (_Float16)(acc[r] + b_qkv[o]);
      if (o < 512) {
        int head = o >> 6, d = o & 63;
        q_t[((size_t)(b * 8 + head) * 1024 + i) * 64 + d] = hv;
      } else if (o < 1024) {
        int oc = o - 512, head = oc >> 6, d = oc & 63;
        k_t[((size_t)(b * 8 + head) * 1024 + i) * 64 + d] = hv;
      } else {
        int oc = o - 1024, head = oc >> 6, d = oc & 63;
        v_m[((size_t)(b * 8 + head) * 64 + d) * 1024 + i] = hv;
      }
    }
  }
}

// ---------------- Kernel 3: attention (16 query rows / block, exact softmax) ----------------
__global__ __launch_bounds__(128) void attn_kernel(const _Float16* __restrict__ q_t,
                                                   const _Float16* __restrict__ k_t,
                                                   const _Float16* __restrict__ v_m,
                                                   _Float16* __restrict__ att_t) {
  extern __shared__ char smem[];
  float*    S = (float*)smem;                                       // [16][1024] f32
  _Float16* P = (_Float16*)(smem + 16 * 1024 * 4);                  // [16][1024] f16
  float*  red = (float*)(smem + 16 * 1024 * 4 + 16 * 1024 * 2);     // [16][8]
  float* rowstat = red + 16 * 8;                                    // [16]

  int t = threadIdx.x;
  int lane = t & 31, wave = t >> 5;
  int lg = lane >> 4, ln = lane & 15;
  int m0 = blockIdx.x * 16;
  int bh = blockIdx.y;
  const _Float16* qbase = q_t + (size_t)bh * 1024 * 64;
  const _Float16* kbase = k_t + (size_t)bh * 1024 * 64;
  const _Float16* vbase = v_m + (size_t)bh * 64 * 1024;
  const float scale = 0.125f; // 1/sqrt(64)

  // Phase 1: S[16][1024] = scale * q^T k ; A-frags hoisted (reused by all 16 col tiles)
  const _Float16* arow = qbase + (size_t)(m0 + ln) * 64;
  v16h aq0 = load_a(arow, 0, lg);
  v16h aq1 = load_a(arow, 32, lg);
  for (int tile = 0; tile < 16; ++tile) {
    int n0 = wave * 256 + tile * 16;
    const _Float16* bcol = kbase + (size_t)(n0 + ln) * 64;
    v8f acc = {};
    acc = WMMA_F16(aq0, load_b(bcol, 0, lg), acc);
    acc = WMMA_F16(aq1, load_b(bcol, 32, lg), acc);
    for (int r = 0; r < 8; ++r) {
      int m = r + lg * 8;
      S[m * 1024 + n0 + ln] = acc[r] * scale;
    }
  }
  __syncthreads();

  // Phase 2: exact row softmax, 8 threads per row
  int row = t >> 3, sub = t & 7;
  float lmax = -3.0e38f;
  for (int c = sub * 128; c < sub * 128 + 128; ++c)
    lmax = fmaxf(lmax, S[row * 1024 + c]);
  red[row * 8 + sub] = lmax;
  __syncthreads();
  if (sub == 0) {
    float m = red[row * 8];
    for (int j = 1; j < 8; ++j) m = fmaxf(m, red[row * 8 + j]);
    rowstat[row] = m;
  }
  __syncthreads();
  float rmax = rowstat[row];
  float lsum = 0.f;
  for (int c = sub * 128; c < sub * 128 + 128; ++c) {
    float e = __expf(S[row * 1024 + c] - rmax);
    S[row * 1024 + c] = e;
    lsum += e;
  }
  red[row * 8 + sub] = lsum;
  __syncthreads();
  if (sub == 0) {
    float s = 0.f;
    for (int j = 0; j < 8; ++j) s += red[row * 8 + j];
    rowstat[row] = 1.0f / s;
  }
  __syncthreads();
  float rinv = rowstat[row];
  for (int c = sub * 128; c < sub * 128 + 128; ++c)
    P[row * 1024 + c] = (_Float16)(S[row * 1024 + c] * rinv);
  __syncthreads();

  // Phase 3: out(16x64) = P(16x1024) @ v^T(1024x64); one 16-col tile per wave
  {
    int n0 = wave * 16;
    const _Float16* parow = P + (size_t)ln * 1024;
    const _Float16* bcol  = vbase + (size_t)(n0 + ln) * 1024;
    v8f acc = {};
    for (int k0 = 0; k0 < 1024; k0 += 32) {
      v16h a  = load_a(parow, k0, lg);
      v16h bb = load_b(bcol, k0, lg);
      acc = WMMA_F16(a, bb, acc);
    }
    int b = bh >> 3, head = bh & 7;
    int d = n0 + ln;
    for (int r = 0; r < 8; ++r) {
      int m = r + lg * 8;
      att_t[((size_t)(b * 1024) + m0 + m) * 512 + head * 64 + d] = (_Float16)acc[r];
    }
  }
}

// ---------------- Kernel 4: output projection + bias + residual, 32x32 tile per wave ----------------
__global__ __launch_bounds__(256) void proj_kernel(const _Float16* __restrict__ wo_h,
                                                   const float* __restrict__ b_out,
                                                   const _Float16* __restrict__ att_t,
                                                   const float* __restrict__ x,
                                                   float* __restrict__ out) {
  int lane = threadIdx.x & 31, wave = threadIdx.x >> 5;
  int g = blockIdx.x * 8 + wave;             // 4096 wave-tiles total
  int b = g >> 9, rem = g & 511;             // 16 x 32 tiles of 32x32 per batch
  int mt = rem >> 5, nt = rem & 31;
  int lg = lane >> 4, ln = lane & 15;
  const _Float16* arow0 = wo_h + (size_t)(mt * 32 + ln) * 512;
  const _Float16* arow1 = arow0 + (size_t)16 * 512;
  const _Float16* bcol0 = att_t + (size_t)(b * 1024 + nt * 32 + ln) * 512;
  const _Float16* bcol1 = bcol0 + (size_t)16 * 512;
  v8f acc00 = {}, acc01 = {}, acc10 = {}, acc11 = {};
  for (int k0 = 0; k0 < 512; k0 += 32) {
    v16h a0 = load_a(arow0, k0, lg);
    v16h a1 = load_a(arow1, k0, lg);
    v16h b0 = load_b(bcol0, k0, lg);
    v16h b1 = load_b(bcol1, k0, lg);
    acc00 = WMMA_F16(a0, b0, acc00);
    acc01 = WMMA_F16(a0, b1, acc01);
    acc10 = WMMA_F16(a1, b0, acc10);
    acc11 = WMMA_F16(a1, b1, acc11);
  }
  for (int half = 0; half < 4; ++half) {
    const v8f& acc = half == 0 ? acc00 : half == 1 ? acc01 : half == 2 ? acc10 : acc11;
    int mo = (half >> 1) * 16, no = (half & 1) * 16;
    int i = nt * 32 + no + ln;
    for (int r = 0; r < 8; ++r) {
      int m = r + lg * 8;
      int o = mt * 32 + mo + m;
      size_t idx = (size_t)(b * 512 + o) * 1024 + i;
      out[idx] = acc[r] + b_out[o] + x[idx];
    }
  }
}

extern "C" void kernel_launch(void* const* d_in, const int* in_sizes, int n_in,
                              void* d_out, int out_size, void* d_ws, size_t ws_size,
                              hipStream_t stream) {
  const float* x     = (const float*)d_in[0];
  const float* gamma = (const float*)d_in[1];
  const float* beta  = (const float*)d_in[2];
  const float* w_qkv = (const float*)d_in[3];
  const float* b_qkv = (const float*)d_in[4];
  const float* w_out = (const float*)d_in[5];
  const float* b_out = (const float*)d_in[6];
  float* out = (float*)d_out;

  char* ws = (char*)d_ws;
  _Float16* wq_h  = (_Float16*)(ws + 0);         // 1536*512*2  = 1572864
  _Float16* wo_h  = (_Float16*)(ws + 1572864);   //  512*512*2  =  524288
  _Float16* xn_t  = (_Float16*)(ws + 2097152);   // 8*1024*512*2 = 8388608
  _Float16* q_t   = (_Float16*)(ws + 10485760);  // 8MB
  _Float16* k_t   = (_Float16*)(ws + 18874368);  // 8MB
  _Float16* v_m   = (_Float16*)(ws + 27262976);  // 8MB
  _Float16* att_t = (_Float16*)(ws + 35651584);  // 8MB  (total ~44MB)

  cvt_kernel<<<4096, 256, 0, stream>>>(w_qkv, w_out, wq_h, wo_h);
  gn_kernel<<<256, 256, 0, stream>>>(x, gamma, beta, xn_t);
  qkv_kernel<<<1536, 256, 0, stream>>>(wq_h, b_qkv, xn_t, q_t, k_t, v_m);
  dim3 ag(64, 64);
  size_t smem = 16 * 1024 * 4 + 16 * 1024 * 2 + (16 * 8 + 16) * 4; // ~96.6 KB dynamic LDS
  attn_kernel<<<ag, 128, smem, stream>>>(q_t, k_t, v_m, att_t);
  proj_kernel<<<512, 256, 0, stream>>>(wo_h, b_out, att_t, x, out);
}
